// RecurrentApproximateBNN_24567212933829
// MI455X (gfx1250) — matrix operand order repaired and checked by reading, not compiled, not run on hardware
//
#include <hip/hip_runtime.h>
#include <hip/hip_bf16.h>

// Problem dims (match reference)
#define BB      64
#define TT      512
#define IN_DIM  512
#define XD      1024
#define OUT_DIM 256

typedef __attribute__((ext_vector_type(16))) __bf16 v16bf;
typedef __attribute__((ext_vector_type(8)))  float  v8f;

// f32 -> bf16 round-to-nearest-even (bit ops only)
__device__ __forceinline__ unsigned short f2bf(float f) {
  unsigned int u = __float_as_uint(f);
  u += 0x7fffu + ((u >> 16) & 1u);
  return (unsigned short)(u >> 16);
}

__device__ __forceinline__ unsigned int f2bf_pk(float lo, float hi) {
  return (unsigned int)f2bf(lo) | ((unsigned int)f2bf(hi) << 16);
}

// Branchless per-dimension random activation + clamp.
// idx is divergent per lane, so a switch would pay all five paths PLUS
// exec-mask branching; instead compute all five (2 shared v_exp_f32 TRANS
// ops that co-execute with VALU) and select with v_cndmask chains.
__device__ __forceinline__ float rand_act1(float v, int id, float mx) {
  const float en    = __expf(-v);                       // shared by sig & tanh
  const float relu  = fmaxf(v, 0.0f);
  const float leaky = fmaxf(v, 0.1f * v);               // exact LeakyReLU(0.1)
  const float sig   = 1.0f / (1.0f + en);
  const float th    = 2.0f / (1.0f + en * en) - 1.0f;   // tanh via e^{-2v}=en^2
  const float selu  = (v > 0.0f) ? 1.0507009873554805f * v
                                 : 1.7580993408473766f * (__expf(v) - 1.0f);
  const float r = (id == 0) ? relu
                : (id == 1) ? sig
                : (id == 2) ? th
                : (id == 3) ? leaky
                : selu;
  return fminf(r, mx);
}

union Frag { v16bf v; int4 q[2]; unsigned short u[16]; };

// One 64-col GEMM tile: C = act(A[64xK] * W[NxK]^T + bias).
// Block = 256 threads = 8 waves; wave w: M tile (w&3)*16, N pair (w>>2)*32.
// nblk selects which 64-column slab of N this block computes.
template<bool AF32>
__device__ __forceinline__ void gemm_tile_body(
    const void* __restrict__ Ap, long lda,
    const unsigned short* __restrict__ W, int K,
    const float* __restrict__ bias, const int* __restrict__ idx,
    const float* __restrict__ mx,
    void* __restrict__ Out, long ldo, int out_f32, int nblk)
{
  const int tid  = threadIdx.x;
  const int wave = tid >> 5;
  const int lane = tid & 31;
  const int h    = lane >> 4;   // 16-lane half
  const int lm   = lane & 15;

  const int mtile  = (wave & 3) << 4;            // 0,16,32,48
  const int ntile0 = nblk * 64 + (wave >> 2) * 32;

  const int am = mtile + lm;    // A row owned by this lane (ISA A-frag layout)

  // W row-major [N,K]: lane lm owns column n; each K-block is two 8-bf16
  // runs (k = 8h.., 16+8h..) per tile => two b128 loads.
  const unsigned short* bp0 = W + (long)(ntile0 + lm) * K + (h << 3);
  const unsigned short* bp1 = W + (long)(ntile0 + 16 + lm) * K + (h << 3);

  v8f c0 = {}; v8f c1 = {};

  for (int kb = 0; kb < K; kb += 32) {
    Frag a;
    if (AF32) {
      // fp32 activations (raw input x): load 16 floats, convert to bf16
      const float* ap0 = (const float*)Ap + (long)am * lda + kb + (h << 3);
      const float* ap1 = ap0 + 16;
      float4 f0 = ((const float4*)ap0)[0];
      float4 f1 = ((const float4*)ap0)[1];
      float4 f2 = ((const float4*)ap1)[0];
      float4 f3 = ((const float4*)ap1)[1];
      a.u[0]=f2bf(f0.x); a.u[1]=f2bf(f0.y); a.u[2]=f2bf(f0.z); a.u[3]=f2bf(f0.w);
      a.u[4]=f2bf(f1.x); a.u[5]=f2bf(f1.y); a.u[6]=f2bf(f1.z); a.u[7]=f2bf(f1.w);
      a.u[8]=f2bf(f2.x); a.u[9]=f2bf(f2.y); a.u[10]=f2bf(f2.z); a.u[11]=f2bf(f2.w);
      a.u[12]=f2bf(f3.x); a.u[13]=f2bf(f3.y); a.u[14]=f2bf(f3.z); a.u[15]=f2bf(f3.w);
    } else {
      // bf16 activations: ISA 16-bit A 16x32 layout -> two b128 loads
      const unsigned short* ap = (const unsigned short*)Ap + (long)am * lda + kb + (h << 3);
      a.q[0] = *(const int4*)ap;
      a.q[1] = *(const int4*)(ap + 16);
    }
    Frag b0, b1;
    b0.q[0] = *(const int4*)(bp0 + kb);
    b0.q[1] = *(const int4*)(bp0 + kb + 16);
    b1.q[0] = *(const int4*)(bp1 + kb);
    b1.q[1] = *(const int4*)(bp1 + kb + 16);

    // Pull next K-block of the weight stream toward the WGP (global_prefetch)
    __builtin_prefetch(bp0 + kb + 32, 0, 3);
    __builtin_prefetch(bp1 + kb + 32, 0, 3);

    c0 = __builtin_amdgcn_wmma_f32_16x16x32_bf16(false, a.v, false, b0.v,
                                                 (short)0, c0, false, false);
    c1 = __builtin_amdgcn_wmma_f32_16x16x32_bf16(false, a.v, false, b1.v,
                                                 (short)0, c1, false, false);
  }

  // Epilogue: N = lane&15 fixed per lane -> one idx/mx/bias load per tile
  const int n0 = ntile0 + lm;
  const int n1 = n0 + 16;
  const float b0s = bias[n0], m0s = mx[n0];
  const float b1s = bias[n1], m1s = mx[n1];
  const int   i0s = idx[n0],  i1s = idx[n1];

#pragma unroll
  for (int r = 0; r < 8; ++r) {
    const int row = mtile + r + (h << 3);   // C/D layout: M = r + 8*half
    const float v0 = rand_act1(c0[r] + b0s, i0s, m0s);
    const float v1 = rand_act1(c1[r] + b1s, i1s, m1s);
    if (out_f32) {
      ((float*)Out)[(long)row * ldo + n0] = v0;
      ((float*)Out)[(long)row * ldo + n1] = v1;
    } else {
      ((unsigned short*)Out)[(long)row * ldo + n0] = f2bf(v0);
      ((unsigned short*)Out)[(long)row * ldo + n1] = f2bf(v1);
    }
  }
}

// Generic single-GEMM kernel; optionally fuses F.normalize(rs) -> h_cat[:,1024:]
template<bool AF32>
__global__ __launch_bounds__(256) void gemm_act_kernel(
    const void* __restrict__ Ap, long lda,
    const unsigned short* __restrict__ W, int K,
    const float* __restrict__ bias, const int* __restrict__ idx,
    const float* __restrict__ mx,
    void* __restrict__ Out, long ldo, int out_f32,
    const float* __restrict__ rs_norm,          // non-null => L2-normalize rs
    unsigned short* __restrict__ norm_dst)      // h_cat + 1024 (row stride 2048)
{
  __shared__ float red[256];

  gemm_tile_body<AF32>(Ap, lda, W, K, bias, idx, mx, Out, ldo, out_f32,
                       (int)blockIdx.x);

  if (rs_norm) {                 // grid must be 16: 16 blocks x 4 rows = 64
    const int tid = threadIdx.x;
    for (int rr = 0; rr < 4; ++rr) {
      const int b = blockIdx.x * 4 + rr;
      float4 v = ((const float4*)(rs_norm + (long)b * XD))[tid];
      red[tid] = v.x * v.x + v.y * v.y + v.z * v.z + v.w * v.w;
      __syncthreads();
      for (int off = 128; off > 0; off >>= 1) {
        if (tid < off) red[tid] += red[tid + off];
        __syncthreads();
      }
      const float s = 1.0f / fmaxf(sqrtf(red[0]), 1e-12f);
      __syncthreads();
      uint2 pk;
      pk.x = f2bf_pk(v.x * s, v.y * s);
      pk.y = f2bf_pk(v.z * s, v.w * s);
      *(uint2*)(norm_dst + (long)b * 2048 + tid * 4) = pk;
      __syncthreads();
    }
  }
}

// Fused dual-head kernel: blocks [0,16) -> recurrent head (Wr -> rs, fp32),
// blocks [16,20) -> output head (Wo -> d_out[:,t,:], fp32). Both read h4.
// Uniform branch on blockIdx: no divergence, one launch instead of two.
__global__ __launch_bounds__(256) void dual_head_kernel(
    const unsigned short* __restrict__ A,       // h4, [64 x 1024] bf16
    const unsigned short* __restrict__ Wr, const float* __restrict__ br,
    const int* __restrict__ idx_r, const float* __restrict__ mx_r,
    float* __restrict__ rs,
    const unsigned short* __restrict__ Wo, const float* __restrict__ bo,
    const int* __restrict__ idx_o, const float* __restrict__ mx_o,
    float* __restrict__ outp, long ldo_o)
{
  if (blockIdx.x < 16) {
    gemm_tile_body<false>(A, XD, Wr, XD, br, idx_r, mx_r,
                          rs, XD, 1, (int)blockIdx.x);
  } else {
    gemm_tile_body<false>(A, XD, Wo, XD, bo, idx_o, mx_o,
                          outp, ldo_o, 1, (int)blockIdx.x - 16);
  }
}

__global__ void f32_to_bf16_kernel(const float* __restrict__ s,
                                   unsigned short* __restrict__ d, int n) {
  int i = blockIdx.x * blockDim.x + threadIdx.x;
  if (i < n) d[i] = f2bf(s[i]);
}

__global__ void zero_f32_kernel(float* __restrict__ p, int n) {
  int i = blockIdx.x * blockDim.x + threadIdx.x;
  if (i < n) p[i] = 0.0f;
}

extern "C" void kernel_launch(void* const* d_in, const int* in_sizes, int n_in,
                              void* d_out, int out_size, void* d_ws, size_t ws_size,
                              hipStream_t stream) {
  (void)in_sizes; (void)n_in; (void)out_size; (void)ws_size;

  const float* x    = (const float*)d_in[0];
  const float* Wi   = (const float*)d_in[1];
  const float* bi   = (const float*)d_in[2];
  const float* Wh1  = (const float*)d_in[3];
  const float* bh1  = (const float*)d_in[4];
  const float* Wh2  = (const float*)d_in[5];
  const float* bh2  = (const float*)d_in[6];
  const float* Wh3  = (const float*)d_in[7];
  const float* bh3  = (const float*)d_in[8];
  const float* Wo   = (const float*)d_in[9];
  const float* bo   = (const float*)d_in[10];
  const float* Wr   = (const float*)d_in[11];
  const float* br   = (const float*)d_in[12];
  const float* max_in  = (const float*)d_in[13];
  const float* max_h1  = (const float*)d_in[14];
  const float* max_h2  = (const float*)d_in[15];
  const float* max_h3  = (const float*)d_in[16];
  const float* max_out = (const float*)d_in[17];
  const float* max_rec = (const float*)d_in[18];
  const int* idx_in  = (const int*)d_in[19];
  const int* idx_h1  = (const int*)d_in[20];
  const int* idx_h2  = (const int*)d_in[21];
  const int* idx_h3  = (const int*)d_in[22];
  const int* idx_out = (const int*)d_in[23];
  const int* idx_rec = (const int*)d_in[24];
  float* out = (float*)d_out;

  // Workspace carve-up (~13 MB; bf16 weights stay L2-resident across steps)
  char* ws = (char*)d_ws;
  auto carve = [&](size_t bytes) {
    void* p = ws;
    ws += (bytes + 255) & ~(size_t)255;
    return p;
  };
  unsigned short* Wibf  = (unsigned short*)carve((size_t)XD * IN_DIM * 2);
  unsigned short* Wh1bf = (unsigned short*)carve((size_t)XD * 2 * XD * 2);
  unsigned short* Wh2bf = (unsigned short*)carve((size_t)XD * XD * 2);
  unsigned short* Wh3bf = (unsigned short*)carve((size_t)XD * XD * 2);
  unsigned short* Wobf  = (unsigned short*)carve((size_t)OUT_DIM * XD * 2);
  unsigned short* Wrbf  = (unsigned short*)carve((size_t)XD * XD * 2);
  unsigned short* hcat  = (unsigned short*)carve((size_t)BB * 2 * XD * 2);
  unsigned short* h2    = (unsigned short*)carve((size_t)BB * XD * 2);
  unsigned short* h3    = (unsigned short*)carve((size_t)BB * XD * 2);
  unsigned short* h4    = (unsigned short*)carve((size_t)BB * XD * 2);
  float*          rs    = (float*)carve((size_t)BB * XD * 4);

  // Per-launch weight conversion (deterministic every call)
  auto cvt = [&](const float* s, unsigned short* d, int n) {
    f32_to_bf16_kernel<<<(n + 255) / 256, 256, 0, stream>>>(s, d, n);
  };
  cvt(Wi,  Wibf,  XD * IN_DIM);
  cvt(Wh1, Wh1bf, XD * 2 * XD);
  cvt(Wh2, Wh2bf, XD * XD);
  cvt(Wh3, Wh3bf, XD * XD);
  cvt(Wo,  Wobf,  OUT_DIM * XD);
  cvt(Wr,  Wrbf,  XD * XD);
  zero_f32_kernel<<<(BB * XD + 255) / 256, 256, 0, stream>>>(rs, BB * XD);

  for (int t = 0; t < TT; ++t) {
    // input GEMM (fp32 A straight from x) + fused normalize(rs) -> h_cat
    gemm_act_kernel<true><<<XD / 64, 256, 0, stream>>>(
        x + (long)t * IN_DIM, (long)TT * IN_DIM, Wibf, IN_DIM,
        bi, idx_in, max_in, hcat, 2 * XD, 0, rs, hcat + XD);
    // hidden1: [64 x 2048] -> [64 x 1024]
    gemm_act_kernel<false><<<XD / 64, 256, 0, stream>>>(
        hcat, 2 * XD, Wh1bf, 2 * XD, bh1, idx_h1, max_h1,
        h2, XD, 0, nullptr, nullptr);
    // hidden2
    gemm_act_kernel<false><<<XD / 64, 256, 0, stream>>>(
        h2, XD, Wh2bf, XD, bh2, idx_h2, max_h2, h3, XD, 0, nullptr, nullptr);
    // hidden3
    gemm_act_kernel<false><<<XD / 64, 256, 0, stream>>>(
        h3, XD, Wh3bf, XD, bh3, idx_h3, max_h3, h4, XD, 0, nullptr, nullptr);
    // fused heads: Wr -> rs (blocks 0..15), Wo -> d_out[:,t,:] (blocks 16..19)
    dual_head_kernel<<<20, 256, 0, stream>>>(
        h4, Wrbf, br, idx_rec, max_rec, rs,
        Wobf, bo, idx_out, max_out,
        out + (long)t * OUT_DIM, (long)TT * OUT_DIM);
  }
}
